// convconv_32538672234774
// MI455X (gfx1250) — matrix operand order repaired and checked by reading, not compiled, not run on hardware
//
#include <hip/hip_runtime.h>

typedef __attribute__((ext_vector_type(16))) _Float16 v16h;
typedef __attribute__((ext_vector_type(8)))  _Float16 v8h;
typedef __attribute__((ext_vector_type(8)))  float    v8f;

#define BATCH 256
#define CH 64

// ---------------------------------------------------------------------------
// Prep: expand grouped [64,2,3,3] f32 weights into dense-within-tile f16
// WMMA A-fragments.  Layout: wfrag[t(4)][kchunk(5)][lane(32)] : v16h
// Tile t: dense 16(co) x 144(k = p*16 + ci_local, p = kh*3+kw), zero-padded
// to K=160 (5 chunks of 32).
// A-fragment slot mapping used here (16-bit A 16x32, contiguous K; B matches):
//   lane = half*16 + m ; slot s -> K = half*16 + s
// ---------------------------------------------------------------------------
__global__ void prep_wfrag_kernel(const float* __restrict__ w,
                                  v16h* __restrict__ wfrag) {
  int i = blockIdx.x * blockDim.x + threadIdx.x;
  if (i >= 4 * 5 * 32) return;
  int lane = i & 31;
  int c    = (i >> 5) % 5;
  int t    = i / (5 * 32);
  int m    = lane & 15;
  int half = lane >> 4;
  v16h out;
#pragma unroll
  for (int s = 0; s < 16; ++s) {
    int K = half * 16 + s;
    int k = 32 * c + K;
    float val = 0.0f;
    if (k < 144) {
      int p  = k >> 4;       // kernel position 0..8
      int cl = k & 15;       // ci within tile
      int kh = p / 3, kw = p % 3;
      if ((cl >> 1) == (m >> 1)) {         // same 2-channel group
        int co = t * 16 + m;
        val = w[(co * 2 + (cl & 1)) * 9 + kh * 3 + kw];
      }
    }
    out[s] = (_Float16)val;
  }
  wfrag[i] = out;
}

// ---------------------------------------------------------------------------
// Grouped 3x3 stride-2 pad-1 conv + bias + ReLU via WMMA implicit GEMM.
// In: [B,64,S,S] (f32 or f16), out: [B,64,S/2,S/2] f16.
// 128-thread block = 4 waves; wave t computes co-tile t (= ci-tile t) for one
// (batch, packed-output-row group).  N packs R output rows of width OUT.
//
// LDS layout is channel-last with zero-padded w columns:
//   lds[(rr*(S+2) + w+1)*CPAD + ci]  (CPAD=72 halfs keeps rows 16B-aligned)
// so one B fragment (16 channels at a fixed kernel position) is a contiguous
// 32B run -> two ds_load_b128, no bounds checks, no EXEC masking.
// The K>=144 tail (p==9) reads from a zeroed LDS block via address select.
// ---------------------------------------------------------------------------
template <int S, bool F32IN>
__global__ void gconv3x3_wmma_kernel(const void* __restrict__ in,
                                     const v16h* __restrict__ wfrag,
                                     const float* __restrict__ bias,
                                     _Float16* __restrict__ out) {
  constexpr int OUT = S / 2;
  constexpr int R = (OUT >= 16) ? 1 : (16 / OUT);  // output rows per tile
  constexpr int NROWS = 2 * R + 1;                 // staged input rows
  constexpr int BLOCKS_PER_B = OUT / R;
  constexpr int CPAD = 72;                         // halfs per (row,w) slot
  constexpr int ZOFF = NROWS * (S + 2) * CPAD;     // zero block (16 halfs)

  const int b   = blockIdx.x / BLOCKS_PER_B;
  const int hb  = (blockIdx.x % BLOCKS_PER_B) * R;  // first output row
  const int tid  = threadIdx.x;
  const int wave = tid >> 5;
  const int lane = tid & 31;

  extern __shared__ _Float16 lds[];

  // ---- zero the w-pad columns and the K-tail zero block ----
  for (int i = tid; i < NROWS * 2 * CH; i += blockDim.x) {
    int ci   = i % CH;
    int rr   = (i / CH) >> 1;
    int side = (i / CH) & 1;
    int wcol = side ? (S + 1) : 0;
    lds[(rr * (S + 2) + wcol) * CPAD + ci] = (_Float16)0.0f;
  }
  if (tid < 16) lds[ZOFF + tid] = (_Float16)0.0f;

  // ---- cooperative stage (global w-fastest: coalesced; LDS channel-last) --
  const int h0 = 2 * hb - 1;
  const int total = CH * NROWS * S;
  for (int i = tid; i < total; i += blockDim.x) {
    int w    = i & (S - 1);
    int rest = i / S;
    int rr   = rest % NROWS;
    int ci   = rest / NROWS;
    int h    = h0 + rr;
    float v  = 0.0f;
    if (h >= 0 && h < S) {
      int idx = ((b * CH + ci) * S + h) * S + w;
      v = F32IN ? ((const float*)in)[idx]
                : (float)((const _Float16*)in)[idx];
    }
    lds[(rr * (S + 2) + w + 1) * CPAD + ci] = (_Float16)v;
  }
  __syncthreads();

  // ---- implicit GEMM: C[16 co, 16 n] += A[16,32] x B[32,16], 5 chunks ----
  const int t     = wave;
  const int n     = lane & 15;
  const int half  = lane >> 4;
  const int r_out = n / OUT;   // packed output row within tile
  const int wo    = n % OUT;

  v8f acc = {};
#pragma unroll
  for (int c = 0; c < 5; ++c) {
    v16h afrag = wfrag[(t * 5 + c) * 32 + lane];

    // B slots s=0..15 of this (chunk, half) = all 16 channels at kernel
    // position p = 2c + half  (p==9 -> zero block).
    const int p   = 2 * c + half;
    const int kh  = p / 3;
    const int kw  = p - 3 * kh;
    const int rr  = 2 * r_out + kh;
    const int win = 2 * wo - 1 + kw;
    int off = (rr * (S + 2) + win + 1) * CPAD + t * 16;
    off = (p <= 8) ? off : ZOFF;               // branch-free tail handling
    v8h blo = *(const v8h*)(lds + off);
    v8h bhi = *(const v8h*)(lds + off + 8);
    v16h bfrag = __builtin_shufflevector(blo, bhi,
        0, 1, 2, 3, 4, 5, 6, 7, 8, 9, 10, 11, 12, 13, 14, 15);

    acc = __builtin_amdgcn_wmma_f32_16x16x32_f16(
        /*neg_a=*/false, afrag, /*neg_b=*/false, bfrag,
        /*c_mod=*/(short)0, acc, /*reuse_a=*/false, /*reuse_b=*/false);
  }

  // ---- bias + ReLU + f16 store (C layout: VGPR r -> M = r + 8*half) ----
  const int ho = hb + r_out;
#pragma unroll
  for (int r = 0; r < 8; ++r) {
    int m  = r + half * 8;
    int co = t * 16 + m;
    float v = acc[r] + bias[co];
    v = v > 0.0f ? v : 0.0f;
    out[((b * CH + co) * OUT + ho) * OUT + wo] = (_Float16)v;
  }
}

// ---------------------------------------------------------------------------
// Conv4: grouped 2x2 s2 p1 on [B,64,4,4] -> sigmoid -> e [B,64,9] f32.
// Tiny (147k outputs): plain VALU.
// ---------------------------------------------------------------------------
__global__ void conv4_sigmoid_kernel(const _Float16* __restrict__ dbuf,
                                     const float* __restrict__ w4,
                                     const float* __restrict__ b4,
                                     float* __restrict__ ebuf) {
  int i = blockIdx.x * blockDim.x + threadIdx.x;
  if (i >= BATCH * CH * 9) return;
  int pos = i % 9;
  int co  = (i / 9) % CH;
  int b   = i / (9 * CH);
  int ho = pos / 3, wo = pos % 3;
  int g0 = co & ~1;  // first input channel of this group
  float acc = b4[co];
#pragma unroll
  for (int cl = 0; cl < 2; ++cl)
#pragma unroll
    for (int kh = 0; kh < 2; ++kh)
#pragma unroll
      for (int kw = 0; kw < 2; ++kw) {
        int h = 2 * ho - 1 + kh;
        int w = 2 * wo - 1 + kw;
        if (h >= 0 && h < 4 && w >= 0 && w < 4) {
          float x = (float)dbuf[((b * CH + g0 + cl) * 4 + h) * 4 + w];
          acc += x * w4[((co * 2 + cl) * 2 + kh) * 2 + kw];
        }
      }
  ebuf[i] = 1.0f / (1.0f + __expf(-acc));
}

// ---------------------------------------------------------------------------
// Dynamic per-(b,c) depthwise 3x3 with reflect padding.  Bandwidth-bound
// streaming pass: one block per (b,c) plane, 32x32 f32 tile staged in LDS
// (float4 loads), 9-tap FMA per pixel, 4 pixels per thread.
// Input plane was pulled into L2 by conv1, so this mostly hits L2.
// ---------------------------------------------------------------------------
__global__ void dyn_depthwise_kernel(const float* __restrict__ a,
                                     const float* __restrict__ ebuf,
                                     float* __restrict__ out) {
  const int plane = blockIdx.x;  // b*64 + c
  __shared__ float tile[32 * 32];
  __shared__ float ke[9];
  const int tid = threadIdx.x;

  const float4* src4 = (const float4*)(a + (size_t)plane * 1024);
  ((float4*)tile)[tid] = src4[tid];  // 256 threads x 16B = 4 KB
  if (tid < 9) ke[tid] = ebuf[plane * 9 + tid];
  __syncthreads();

  const int r     = tid >> 3;        // row 0..31
  const int cbase = (tid & 7) * 4;   // 4 consecutive cols
#pragma unroll
  for (int j = 0; j < 4; ++j) {
    const int col = cbase + j;
    float s = 0.0f;
#pragma unroll
    for (int ki = 0; ki < 3; ++ki) {
      int hh = r - 1 + ki;
      hh = hh < 0 ? -hh : (hh > 31 ? 62 - hh : hh);  // reflect (no edge dup)
#pragma unroll
      for (int kj = 0; kj < 3; ++kj) {
        int ww = col - 1 + kj;
        ww = ww < 0 ? -ww : (ww > 31 ? 62 - ww : ww);
        s += tile[hh * 32 + ww] * ke[ki * 3 + kj];
      }
    }
    out[(size_t)plane * 1024 + r * 32 + col] = s;
  }
}

// ---------------------------------------------------------------------------
// Launcher.  Workspace layout (bytes):
//   [0)        wfrag1 (20480)
//   [20480)    wfrag2 (20480)
//   [40960)    wfrag3 (20480)
//   [61440)    bbuf f16 [256,64,16,16]  (8388608)
//   [8450048)  cbuf f16 [256,64,8,8]    (2097152)
//   [10547200) dbuf f16 [256,64,4,4]    (524288)
//   [11071488) ebuf f32 [256,64,9]      (589824)   total ~11.7 MB
// ---------------------------------------------------------------------------
extern "C" void kernel_launch(void* const* d_in, const int* in_sizes, int n_in,
                              void* d_out, int out_size, void* d_ws, size_t ws_size,
                              hipStream_t stream) {
  const float* a  = (const float*)d_in[0];
  const float* w1 = (const float*)d_in[1];
  const float* b1 = (const float*)d_in[2];
  const float* w2 = (const float*)d_in[3];
  const float* b2 = (const float*)d_in[4];
  const float* w3 = (const float*)d_in[5];
  const float* b3 = (const float*)d_in[6];
  const float* w4 = (const float*)d_in[7];
  const float* b4 = (const float*)d_in[8];

  char* ws = (char*)d_ws;
  v16h*      wf1  = (v16h*)(ws + 0);
  v16h*      wf2  = (v16h*)(ws + 20480);
  v16h*      wf3  = (v16h*)(ws + 40960);
  _Float16*  bbuf = (_Float16*)(ws + 61440);
  _Float16*  cbuf = (_Float16*)(ws + 8450048);
  _Float16*  dbuf = (_Float16*)(ws + 10547200);
  float*     ebuf = (float*)(ws + 11071488);
  float*     xout = (float*)d_out;

  // weight fragment expansion (tiny)
  prep_wfrag_kernel<<<5, 128, 0, stream>>>(w1, wf1);
  prep_wfrag_kernel<<<5, 128, 0, stream>>>(w2, wf2);
  prep_wfrag_kernel<<<5, 128, 0, stream>>>(w3, wf3);

  // conv stack (WMMA implicit GEMM), intermediates in f16
  // dynamic LDS bytes = (NROWS*(S+2)*72 + 16) * 2
  gconv3x3_wmma_kernel<32, true ><<<BATCH * 16, 128, (3 * 34 * 72 + 16) * 2, stream>>>(
      (const void*)a, wf1, b1, bbuf);
  gconv3x3_wmma_kernel<16, false><<<BATCH * 4, 128, (5 * 18 * 72 + 16) * 2, stream>>>(
      (const void*)bbuf, wf2, b2, cbuf);
  gconv3x3_wmma_kernel<8,  false><<<BATCH * 1, 128, (9 * 10 * 72 + 16) * 2, stream>>>(
      (const void*)cbuf, wf3, b3, dbuf);

  // conv4 + sigmoid -> dynamic kernels e
  conv4_sigmoid_kernel<<<(BATCH * CH * 9 + 255) / 256, 256, 0, stream>>>(
      dbuf, w4, b4, ebuf);

  // dynamic depthwise 3x3 with reflect pad (bandwidth-bound streaming pass)
  dyn_depthwise_kernel<<<BATCH * CH, 256, 0, stream>>>(a, ebuf, xout);
}
